// Cross_Attention_13314398617641
// MI455X (gfx1250) — compile-verified, matrix operand order checked by reading, not run on hardware
//
#include <hip/hip_runtime.h>
#include <hip/hip_bf16.h>
#include <math.h>

// ---- problem constants --------------------------------------------------
#define SRC_LEN 4096
#define BATCH   64
#define DH      256   // DEC_HID (N dimension of big GEMM)
#define EH2     512   // 2*ENC_HID (K dimension of big GEMM)
#define WCOLS   768   // W row length

typedef __attribute__((ext_vector_type(16))) __bf16 v16bf;
typedef __attribute__((ext_vector_type(8)))  __bf16 v8bf;
typedef __attribute__((ext_vector_type(8)))  float  v8f;
typedef __attribute__((ext_vector_type(4)))  float  v4f;

union V16U { v16bf v; __bf16 e[16]; };

// ---- prep 1: We (W[:,256:768]) fp32 -> bf16 workspace -------------------
__global__ __launch_bounds__(256) void prep_we_bf16(const float* __restrict__ W,
                                                    __bf16* __restrict__ weBF) {
    int idx = blockIdx.x * 256 + threadIdx.x;   // 0 .. 256*512-1
    int k = idx >> 9;          // row (0..255)
    int e = idx & 511;         // col (0..511)
    weBF[idx] = (__bf16)W[k * WCOLS + DH + e];
}

// ---- prep 2: hWhb[b][k] = h[b,:] . Wh[k,:] + bias[k] --------------------
__global__ __launch_bounds__(256) void prep_hwhb(const float* __restrict__ hidden,
                                                 const float* __restrict__ W,
                                                 const float* __restrict__ bias,
                                                 float* __restrict__ hWhb) {
    int b = blockIdx.x;        // 0..63
    int k = threadIdx.x;       // 0..255
    const float* hrow = hidden + b * DH;
    const float* wrow = W + (size_t)k * WCOLS;   // Wh = W[k][0:256]
    float s = bias[k];
    #pragma unroll 8
    for (int d = 0; d < DH; ++d) s += hrow[d] * wrow[d];
    hWhb[b * DH + k] = s;
}

// ---- main: scores[b][s] = v . tanh(eo[b,s,:]@We^T + hWhb[b,:]) ----------
// One wave handles M=32 src rows (2 row tiles sharing each B fragment) x
// full N=256; K=512 in 16 chunks of 32. B fragments are reused by 2 WMMAs,
// halving redundant L2 traffic for We.
__global__ __launch_bounds__(256, 1) void attn_scores(
        const float*  __restrict__ eo,     // (4096,64,512)
        const __bf16* __restrict__ weBF,   // (256,512) bf16
        const float*  __restrict__ hWhb,   // (64,256)
        const float*  __restrict__ vvec,   // (256)
        float*        __restrict__ scores) // (64,4096)
{
    const int wave = threadIdx.x >> 5;
    const int lane = threadIdx.x & 31;
    const int tile = blockIdx.x * 8 + wave;     // 0..127 (src tiles of 32)
    const int b    = blockIdx.y;                // 0..63
    const int s0   = tile * 32;
    const int col  = lane & 15;
    const bool hi  = lane >= 16;

    v8f zero = {0.f, 0.f, 0.f, 0.f, 0.f, 0.f, 0.f, 0.f};
    v8f acc[2][16];
    #pragma unroll
    for (int g = 0; g < 2; ++g)
        #pragma unroll
        for (int t = 0; t < 16; ++t) acc[g][t] = zero;

    // A-fragment source row for this lane (ISA 16-bit A layout: lanes 0-15
    // hold M=0..15 with K {0..7,16..23}; lanes 16-31 same M, K {8..15,24..31}).
    // Row tile g=1 is a constant +16 rows = 2 MB offset (fits imm24).
    const float* arow = eo + ((size_t)(s0 + col) * BATCH + b) * EH2;
    const size_t gstep = (size_t)16 * BATCH * EH2;   // 16 rows ahead
    const int h8 = hi ? 8 : 0;

    #pragma unroll 1
    for (int c = 0; c < 16; ++c) {              // K chunks of 32
        const int e0 = c * 32 + h8;             // first 8 K-values for this lane
        const int e1 = c * 32 + 16 + h8;        // second 8 K-values

        V16U a[2];
        #pragma unroll
        for (int g = 0; g < 2; ++g) {
            const float* ar = arow + g * gstep;
            v4f f0 = *(const v4f*)(ar + e0);
            v4f f1 = *(const v4f*)(ar + e0 + 4);
            v4f f2 = *(const v4f*)(ar + e1);
            v4f f3 = *(const v4f*)(ar + e1 + 4);
            #pragma unroll
            for (int i = 0; i < 4; ++i) {
                a[g].e[i]      = (__bf16)f0[i];
                a[g].e[4 + i]  = (__bf16)f1[i];
                a[g].e[8 + i]  = (__bf16)f2[i];
                a[g].e[12 + i] = (__bf16)f3[i];
            }
        }

        #pragma unroll
        for (int t = 0; t < 16; ++t) {          // N tiles (16 cols each)
            // B-fragment: lane n holds column N=n (k = 16t+n), mirrored K split
            const __bf16* brow = weBF + (size_t)(t * 16 + col) * EH2;
            v8bf b0 = *(const v8bf*)(brow + e0);
            v8bf b1 = *(const v8bf*)(brow + e1);
            V16U bf;
            bf.v = __builtin_shufflevector(b0, b1,
                     0, 1, 2, 3, 4, 5, 6, 7, 8, 9, 10, 11, 12, 13, 14, 15);
            acc[0][t] = __builtin_amdgcn_wmma_f32_16x16x32_bf16(
                            false, a[0].v, false, bf.v, (short)0, acc[0][t], false, false);
            acc[1][t] = __builtin_amdgcn_wmma_f32_16x16x32_bf16(
                            false, a[1].v, false, bf.v, (short)0, acc[1][t], false, false);
        }
    }

    // Epilogue: C layout — lane L, VGPR r => row M = r + (L>=16 ? 8 : 0),
    // col N = L&15. Apply tanh + v-dot, reduce N across 16-lane halves.
    float part[2][8];
    #pragma unroll
    for (int g = 0; g < 2; ++g)
        #pragma unroll
        for (int r = 0; r < 8; ++r) part[g][r] = 0.f;

    #pragma unroll
    for (int t = 0; t < 16; ++t) {
        int k = t * 16 + col;
        float add = hWhb[b * DH + k];
        float vk  = vvec[k];
        #pragma unroll
        for (int g = 0; g < 2; ++g)
            #pragma unroll
            for (int r = 0; r < 8; ++r)
                part[g][r] += vk * tanhf(acc[g][t][r] + add);
    }

    #pragma unroll
    for (int m = 1; m < 16; m <<= 1) {
        #pragma unroll
        for (int g = 0; g < 2; ++g)
            #pragma unroll
            for (int r = 0; r < 8; ++r)
                part[g][r] += __shfl_xor(part[g][r], m, 32);
    }

    if (col == 0) {                              // lanes 0 (rows 0-7) and 16 (rows 8-15)
        const int rbase = hi ? 8 : 0;
        #pragma unroll
        for (int g = 0; g < 2; ++g)
            #pragma unroll
            for (int r = 0; r < 8; ++r)
                scores[(size_t)b * SRC_LEN + s0 + g * 16 + rbase + r] = part[g][r];
    }
}

// ---- softmax over s (4096) per batch row, in place in d_out -------------
__global__ __launch_bounds__(256) void softmax_rows(float* __restrict__ out) {
    __shared__ float red[256];
    const int b = blockIdx.x;
    const int t = threadIdx.x;
    float* row = out + (size_t)b * SRC_LEN;

    float x[16];
    #pragma unroll
    for (int i = 0; i < 16; ++i) x[i] = row[t + i * 256];

    float m = -INFINITY;
    #pragma unroll
    for (int i = 0; i < 16; ++i) m = fmaxf(m, x[i]);
    red[t] = m; __syncthreads();
    for (int s = 128; s > 0; s >>= 1) {
        if (t < s) red[t] = fmaxf(red[t], red[t + s]);
        __syncthreads();
    }
    m = red[0]; __syncthreads();

    float sum = 0.f;
    #pragma unroll
    for (int i = 0; i < 16; ++i) { x[i] = __expf(x[i] - m); sum += x[i]; }
    red[t] = sum; __syncthreads();
    for (int s = 128; s > 0; s >>= 1) {
        if (t < s) red[t] += red[t + s];
        __syncthreads();
    }
    float inv = 1.f / red[0];
    #pragma unroll
    for (int i = 0; i < 16; ++i) row[t + i * 256] = x[i] * inv;
}

// ---- launcher -----------------------------------------------------------
extern "C" void kernel_launch(void* const* d_in, const int* in_sizes, int n_in,
                              void* d_out, int out_size, void* d_ws, size_t ws_size,
                              hipStream_t stream) {
    (void)in_sizes; (void)n_in; (void)out_size; (void)ws_size;
    const float* hidden = (const float*)d_in[0];   // (1,64,256)
    const float* eo     = (const float*)d_in[1];   // (4096,64,512)
    const float* W      = (const float*)d_in[2];   // (256,768)
    const float* bias   = (const float*)d_in[3];   // (256)
    const float* vvec   = (const float*)d_in[4];   // (256)
    float* out = (float*)d_out;                    // (64,4096)

    __bf16* weBF = (__bf16*)d_ws;                              // 256 KB
    float*  hWhb = (float*)((char*)d_ws + 256u * 1024u);       // 64 KB

    prep_we_bf16<<<dim3(512), dim3(256), 0, stream>>>(W, weBF);
    prep_hwhb  <<<dim3(64),  dim3(256), 0, stream>>>(hidden, W, bias, hWhb);
    attn_scores<<<dim3(16, 64), dim3(256), 0, stream>>>(eo, weBF, hWhb, vvec, out);
    softmax_rows<<<dim3(64), dim3(256), 0, stream>>>(out);
}
